// SKAModule_61512521614000
// MI455X (gfx1250) — matrix operand order repaired and checked by reading, not compiled, not run on hardware
//
#include <hip/hip_runtime.h>
#include <hip/hip_bf16.h>
#include <math.h>

typedef __attribute__((ext_vector_type(16))) __bf16 v16bf;
typedef __attribute__((ext_vector_type(8)))  __bf16 v8bf;
typedef __attribute__((ext_vector_type(8)))  float  v8f;
typedef __attribute__((ext_vector_type(2)))  float  v2f;

#define DM 1024   // D_MODEL
#define NH 16     // N_HEADS
#define DH 64     // D_HEAD
#define RK 32     // RANK
#define BB 4
#define TT 4096
#define NC 2048   // packed cols: [0,512)=z  [512,1024)=zq  [1024,2048)=v
#define MROWS (BB*TT)
#define HTOT ((size_t)MROWS * DM)      // 16,777,216 elements of h
#define WTOT ((size_t)NC * DM)         //  2,097,152 elements of packed W

// ---------------------------------------------------------------------------
// K0: one-shot fp32 -> bf16 conversion of h and packed [Wk;Wq;Wv]
// ---------------------------------------------------------------------------
__global__ __launch_bounds__(256)
void k0_cvt(const float* __restrict__ h,  const float* __restrict__ Wk,
            const float* __restrict__ Wq, const float* __restrict__ Wv,
            __bf16* __restrict__ hb, __bf16* __restrict__ wb)
{
  const size_t i0 = ((size_t)blockIdx.x * 256 + threadIdx.x) * 8;
  const float* src;
  __bf16* dst;
  if (i0 < HTOT) {
    src = h + i0;  dst = hb + i0;
  } else {
    size_t j = i0 - HTOT;
    if (j < 524288)       src = Wk + j;
    else if (j < 1048576) src = Wq + (j - 524288);
    else                  src = Wv + (j - 1048576);
    dst = wb + j;
  }
  const float4 f0 = ((const float4*)src)[0];
  const float4 f1 = ((const float4*)src)[1];
  dst[0] = (__bf16)f0.x; dst[1] = (__bf16)f0.y;
  dst[2] = (__bf16)f0.z; dst[3] = (__bf16)f0.w;
  dst[4] = (__bf16)f1.x; dst[5] = (__bf16)f1.y;
  dst[6] = (__bf16)f1.z; dst[7] = (__bf16)f1.w;
}

// ---------------------------------------------------------------------------
// K1: z_all = h @ [Wk;Wq;Wv]^T, LDS-tiled bf16 WMMA GEMM.
// Block tile 64(M) x 128(N), 8 waves as 2x4, each wave 32x32 (4 WMMA accs).
// Double-buffered GLOBAL_LOAD_ASYNC_TO_LDS_B128 pipeline: two tiles in
// flight, s_wait_asynccnt 3 retires only the older tile (3 asyncs/thread).
// ---------------------------------------------------------------------------
#define Bb_M 64
#define Bb_N 128
#define Bb_K 32
#define LPAD 40   // padded LDS row (bf16): 80 B = 5*16 B (keeps 16B alignment)

__global__ __launch_bounds__(256)
void k1_gemm(const __bf16* __restrict__ hb, const __bf16* __restrict__ wb,
             float* __restrict__ zall)
{
  __shared__ __bf16 Abuf[2][Bb_M][LPAD];
  __shared__ __bf16 Bbuf[2][Bb_N][LPAD];

  const int tid  = threadIdx.x;
  const int lane = tid & 31;
  const int w    = tid >> 5;
  const int wm   = w >> 2;                 // 0..1
  const int wn   = w & 3;                  // 0..3
  const int mblk = blockIdx.x * Bb_M;
  const int nblk = blockIdx.y * Bb_N;

  // tile-copy ownership: 16B (8 bf16) chunks per thread
  const int arow = tid >> 2, aseg = tid & 3;          // A: 64 rows x 4 segs
  const int brow = tid >> 2, bseg = tid & 3;          // B: rows brow, brow+64

  const __bf16* gAr  = hb + (size_t)(mblk + arow) * DM + aseg * 8;
  const __bf16* gB0r = wb + (size_t)(nblk + brow) * DM + bseg * 8;
  const __bf16* gB1r = wb + (size_t)(nblk + brow + 64) * DM + bseg * 8;

  const unsigned ldsA0  = (unsigned)(uintptr_t)&Abuf[0][arow][aseg * 8];
  const unsigned ldsA1  = (unsigned)(uintptr_t)&Abuf[1][arow][aseg * 8];
  const unsigned ldsB00 = (unsigned)(uintptr_t)&Bbuf[0][brow][bseg * 8];
  const unsigned ldsB01 = (unsigned)(uintptr_t)&Bbuf[1][brow][bseg * 8];
  const unsigned ldsB10 = (unsigned)(uintptr_t)&Bbuf[0][brow + 64][bseg * 8];
  const unsigned ldsB11 = (unsigned)(uintptr_t)&Bbuf[1][brow + 64][bseg * 8];

#define ISSUE(kkv, pv) do {                                                      \
    asm volatile("global_load_async_to_lds_b128 %0, %1, off"                     \
        :: "v"((pv) ? ldsA1 : ldsA0),                                            \
           "v"((unsigned long long)(uintptr_t)(gAr  + (kkv))) : "memory");       \
    asm volatile("global_load_async_to_lds_b128 %0, %1, off"                     \
        :: "v"((pv) ? ldsB01 : ldsB00),                                          \
           "v"((unsigned long long)(uintptr_t)(gB0r + (kkv))) : "memory");       \
    asm volatile("global_load_async_to_lds_b128 %0, %1, off"                     \
        :: "v"((pv) ? ldsB11 : ldsB10),                                          \
           "v"((unsigned long long)(uintptr_t)(gB1r + (kkv))) : "memory");       \
  } while (0)

  ISSUE(0, 0);
  ISSUE(Bb_K, 1);

  const int aOff = (lane >> 4) * 8;    // A frag K runs [aOff..+7] & [aOff+16..+23]
  const int bOff = (lane >> 4) * 8;    // B frag K runs [bOff..+7] & [bOff+16..+23]
  const int r0 = wm * 32 + (lane & 15);
  const int c0 = wn * 32 + (lane & 15);

  v8f acc00 = {}, acc01 = {}, acc10 = {}, acc11 = {};

  int p = 0;
  for (int kk = 0; kk < DM; kk += Bb_K) {
    if (kk + Bb_K < DM) asm volatile("s_wait_asynccnt 0x3" ::: "memory");
    else                asm volatile("s_wait_asynccnt 0x0" ::: "memory");
    __syncthreads();

    const __bf16 (*Ab)[LPAD] = Abuf[p];
    const __bf16 (*Bbf)[LPAD] = Bbuf[p];
    v16bf a0, a1, b0, b1;
    {
      const v8bf* p0 = (const v8bf*)&Ab[r0][aOff];
      const v8bf* p1 = (const v8bf*)&Ab[r0][aOff + 16];
      a0 = __builtin_shufflevector(*p0, *p1, 0,1,2,3,4,5,6,7,8,9,10,11,12,13,14,15);
      const v8bf* p2 = (const v8bf*)&Ab[r0 + 16][aOff];
      const v8bf* p3 = (const v8bf*)&Ab[r0 + 16][aOff + 16];
      a1 = __builtin_shufflevector(*p2, *p3, 0,1,2,3,4,5,6,7,8,9,10,11,12,13,14,15);
      const v8bf* q0 = (const v8bf*)&Bbf[c0][bOff];
      const v8bf* q1 = (const v8bf*)&Bbf[c0][bOff + 16];
      b0 = __builtin_shufflevector(*q0, *q1, 0,1,2,3,4,5,6,7,8,9,10,11,12,13,14,15);
      const v8bf* q2 = (const v8bf*)&Bbf[c0 + 16][bOff];
      const v8bf* q3 = (const v8bf*)&Bbf[c0 + 16][bOff + 16];
      b1 = __builtin_shufflevector(*q2, *q3, 0,1,2,3,4,5,6,7,8,9,10,11,12,13,14,15);
    }
    acc00 = __builtin_amdgcn_wmma_f32_16x16x32_bf16(false, a0, false, b0, (short)0, acc00, false, false);
    acc01 = __builtin_amdgcn_wmma_f32_16x16x32_bf16(false, a0, false, b1, (short)0, acc01, false, false);
    acc10 = __builtin_amdgcn_wmma_f32_16x16x32_bf16(false, a1, false, b0, (short)0, acc10, false, false);
    acc11 = __builtin_amdgcn_wmma_f32_16x16x32_bf16(false, a1, false, b1, (short)0, acc11, false, false);

    __syncthreads();                       // everyone done reading buf[p]
    if (kk + 2 * Bb_K < DM) ISSUE(kk + 2 * Bb_K, p);
    p ^= 1;
  }
#undef ISSUE

  const int rbase = mblk + wm * 32 + (lane >> 4) * 8;
  const int cbase = nblk + wn * 32 + (lane & 15);
#pragma unroll
  for (int i = 0; i < 8; ++i) {
    zall[(size_t)(rbase + i)      * NC + cbase]      = acc00[i];
    zall[(size_t)(rbase + i)      * NC + cbase + 16] = acc01[i];
    zall[(size_t)(rbase + 16 + i) * NC + cbase]      = acc10[i];
    zall[(size_t)(rbase + 16 + i) * NC + cbase + 16] = acc11[i];
  }
}

// ---------------------------------------------------------------------------
// K2: inv_norm[b,h] = 1 / max(max_t ||z[b,t,h,:]||, 1e-6)
// ---------------------------------------------------------------------------
__global__ __launch_bounds__(256)
void k2_maxnorm(const float* __restrict__ zall, float* __restrict__ invn)
{
  __shared__ float red[256];
  const int bh = blockIdx.x, b = bh >> 4, hh = bh & 15;
  float mx = 0.f;
  for (int t = threadIdx.x; t < TT; t += 256) {
    const float* zp = zall + (size_t)(b * TT + t) * NC + hh * RK;
    float s = 0.f;
#pragma unroll
    for (int r = 0; r < RK; ++r) s += zp[r] * zp[r];
    mx = fmaxf(mx, s);
  }
  red[threadIdx.x] = mx;
  __syncthreads();
  for (int s = 128; s > 0; s >>= 1) {
    if (threadIdx.x < s) red[threadIdx.x] = fmaxf(red[threadIdx.x], red[threadIdx.x + s]);
    __syncthreads();
  }
  if (threadIdx.x == 0)
    invn[bh] = 1.f / fmaxf(sqrtf(red[0]), 1e-6f);
}

// ---------------------------------------------------------------------------
// K3: per-(b,h)  G = Zm^T Zm,  M = (Z_t m)(Z_{t-1} m)^T,  C = (Vm)^T (Zm)
// ---------------------------------------------------------------------------
#define CH 64
__global__ __launch_bounds__(256)
void k3_gram(const float* __restrict__ zall,
             const int* __restrict__ mask,
             const float* __restrict__ invn,
             float* __restrict__ gG, float* __restrict__ gM, float* __restrict__ gC)
{
  __shared__ float zsh[CH][RK];
  __shared__ float vsh[CH][DH];
  __shared__ float msh[CH];
  __shared__ float zprev[RK];
  __shared__ float mprev;

  const int bh = blockIdx.x, b = bh >> 4, hh = bh & 15;
  const int tid = threadIdx.x;
  const float s = invn[bh];

  const int gr = tid >> 3, gs0 = (tid & 7) * 4;
  const int cp = tid >> 2, cr0 = (tid & 3) * 8;

  float gA[4] = {0, 0, 0, 0}, mA[4] = {0, 0, 0, 0};
  float cA[8] = {0, 0, 0, 0, 0, 0, 0, 0};

  if (tid == 0) mprev = 0.f;
  if (tid < RK) zprev[tid] = 0.f;

  for (int t0 = 0; t0 < TT; t0 += CH) {
    __syncthreads();
#pragma unroll
    for (int j = 0; j < (CH * RK) / 256; ++j) {
      int idx = tid + j * 256, row = idx >> 5, c = idx & 31;
      zsh[row][c] = zall[(size_t)(b * TT + t0 + row) * NC + hh * RK + c] * s;
    }
#pragma unroll
    for (int j = 0; j < (CH * DH) / 256; ++j) {
      int idx = tid + j * 256, row = idx >> 6, c = idx & 63;
      vsh[row][c] = zall[(size_t)(b * TT + t0 + row) * NC + 1024 + hh * DH + c];
    }
    if (tid < CH) msh[tid] = (float)mask[b * TT + t0 + tid];
    __syncthreads();

    for (int tt = 0; tt < CH; ++tt) {
      const float m  = msh[tt];
      const float ml = m * ((tt > 0) ? msh[tt - 1] : mprev);
      const float* zc  = zsh[tt];
      const float* zm1 = (tt > 0) ? zsh[tt - 1] : zprev;
      const float zr = zc[gr];
#pragma unroll
      for (int j = 0; j < 4; ++j) {
        gA[j] += m  * zr * zc[gs0 + j];
        mA[j] += ml * zr * zm1[gs0 + j];
      }
      const float vp = vsh[tt][cp];
#pragma unroll
      for (int j = 0; j < 8; ++j)
        cA[j] += m * vp * zc[cr0 + j];
    }
    __syncthreads();
    if (tid < RK)  zprev[tid] = zsh[CH - 1][tid];
    if (tid == RK) mprev = msh[CH - 1];
  }
#pragma unroll
  for (int j = 0; j < 4; ++j) {
    gG[(size_t)bh * 1024 + gr * RK + gs0 + j] = gA[j];
    gM[(size_t)bh * 1024 + gr * RK + gs0 + j] = mA[j];
  }
#pragma unroll
  for (int j = 0; j < 8; ++j)
    gC[(size_t)bh * 2048 + cp * RK + cr0 + j] = cA[j];
}

// ---------------------------------------------------------------------------
// K4: per-(b,h) small linear algebra -> E = eta * B_v * L * A_filt * L^-1 (64x32)
// ---------------------------------------------------------------------------
__global__ __launch_bounds__(64)
void k4_solve(const float* __restrict__ gG, const float* __restrict__ gM,
              const float* __restrict__ gC, const float* __restrict__ eta_p,
              const float* __restrict__ gamma_p, float* __restrict__ gE)
{
  __shared__ float Gs[RK][RK + 1], Ls[RK][RK + 1], Li[RK][RK + 1];
  __shared__ float Ms[RK][RK + 1], T1[RK][RK + 1], T2[RK][RK + 1];
  __shared__ float Aw[RK][RK + 1], Af[RK][RK + 1], Gi[RK][RK + 1];
  __shared__ float Cv[DH][RK], Bv[DH][RK + 1];
  __shared__ float vv[RK], uu[RK], tv[RK];
  __shared__ float snorm, sc2;

  const int bh  = blockIdx.x;
  const int tid = threadIdx.x;
  const float eta = *eta_p;

  for (int j = 0; j < 16; ++j) {
    int idx = tid + j * 64, r = idx >> 5, c = idx & 31;
    float g = 0.5f * (gG[(size_t)bh * 1024 + r * RK + c] +
                      gG[(size_t)bh * 1024 + c * RK + r]);
    Gs[r][c] = g + ((r == c) ? 1e-3f : 0.f);
    Ms[r][c] = gM[(size_t)bh * 1024 + r * RK + c];
    Ls[r][c] = 0.f;
  }
  for (int j = 0; j < 32; ++j) {
    int idx = tid + j * 64, p = idx >> 5, r = idx & 31;
    Cv[p][r] = gC[(size_t)bh * 2048 + p * RK + r];
  }
  __syncthreads();

  for (int j = 0; j < RK; ++j) {                          // Cholesky
    if (tid == 0) {
      float s = Gs[j][j];
      for (int k = 0; k < j; ++k) s -= Ls[j][k] * Ls[j][k];
      Ls[j][j] = sqrtf(fmaxf(s, 1e-20f));
    }
    __syncthreads();
    if (tid > j && tid < RK) {
      float s = Gs[tid][j];
      for (int k = 0; k < j; ++k) s -= Ls[tid][k] * Ls[j][k];
      Ls[tid][j] = s / Ls[j][j];
    }
    __syncthreads();
  }

  if (tid < RK) {                                         // Li = L^-1
    const int c = tid;
    for (int j = 0; j < RK; ++j) {
      float x = 0.f;
      if (j >= c) {
        x = (j == c) ? 1.f : 0.f;
        for (int k = c; k < j; ++k) x -= Ls[j][k] * Li[k][c];
        x /= Ls[j][j];
      }
      Li[j][c] = x;
    }
  }
  __syncthreads();

  for (int j = 0; j < 16; ++j) {                          // T1 = Li * M
    int idx = tid + j * 64, r = idx >> 5, c = idx & 31;
    float a = 0.f;
    for (int k = 0; k < RK; ++k) a += Li[r][k] * Ms[k][c];
    T1[r][c] = a;
  }
  __syncthreads();
  for (int j = 0; j < 16; ++j) {                          // Aw = T1 * Li^T
    int idx = tid + j * 64, r = idx >> 5, c = idx & 31;
    float a = 0.f;
    for (int k = 0; k < RK; ++k) a += T1[r][k] * Li[c][k];
    Aw[r][c] = a;
  }
  __syncthreads();

  if (tid < RK) vv[tid] = 0.17677669529663689f;           // 1/sqrt(32)
  __syncthreads();
  for (int it = 0; it < 6; ++it) {                        // power iteration
    if (tid < RK) { float a = 0; for (int k = 0; k < RK; ++k) a += Aw[tid][k] * vv[k]; uu[tid] = a; }
    __syncthreads();
    if (tid == 0) { float n = 0; for (int k = 0; k < RK; ++k) n += uu[k] * uu[k]; snorm = fmaxf(sqrtf(n), 1e-8f); }
    __syncthreads();
    if (tid < RK) uu[tid] = uu[tid] / snorm;
    __syncthreads();
    if (tid < RK) { float a = 0; for (int k = 0; k < RK; ++k) a += Aw[k][tid] * uu[k]; tv[tid] = a; }
    __syncthreads();
    if (tid == 0) { float n = 0; for (int k = 0; k < RK; ++k) n += tv[k] * tv[k]; snorm = fmaxf(sqrtf(n), 1e-8f); }
    __syncthreads();
    if (tid < RK) vv[tid] = tv[tid] / snorm;
    __syncthreads();
  }
  if (tid < RK) { float a = 0; for (int k = 0; k < RK; ++k) a += Aw[tid][k] * vv[k]; uu[tid] = a; }
  __syncthreads();
  if (tid == 0) {
    float n = 0; for (int k = 0; k < RK; ++k) n += uu[k] * uu[k];
    float scale = fmaxf(sqrtf(n), 1.f);
    float g2 = fminf(fmaxf(*gamma_p, 1.f), 1.5f);
    sc2 = g2 / scale;
  }
  __syncthreads();
  for (int j = 0; j < 16; ++j) {
    int idx = tid + j * 64, r = idx >> 5, c = idx & 31;
    Aw[r][c] *= sc2;
  }
  __syncthreads();

  for (int j = 0; j < 16; ++j) {                          // Af = Aw*Aw ; Gi = Li^T*Li
    int idx = tid + j * 64, r = idx >> 5, c = idx & 31;
    float a = 0.f, g = 0.f;
    for (int k = 0; k < RK; ++k) { a += Aw[r][k] * Aw[k][c]; g += Li[k][r] * Li[k][c]; }
    Af[r][c] = a;  Gi[r][c] = g;
  }
  __syncthreads();

  for (int j = 0; j < 32; ++j) {                          // Bv = Cv * Gi
    int idx = tid + j * 64, p = idx >> 5, r = idx & 31;
    float a = 0.f;
    for (int k = 0; k < RK; ++k) a += Cv[p][k] * Gi[k][r];
    Bv[p][r] = a;
  }
  __syncthreads();

  for (int j = 0; j < 16; ++j) {                          // T2 = L * Af
    int idx = tid + j * 64, r = idx >> 5, c = idx & 31;
    float a = 0.f;
    for (int k = 0; k < RK; ++k) a += Ls[r][k] * Af[k][c];
    T2[r][c] = a;
  }
  __syncthreads();
  for (int j = 0; j < 16; ++j) {                          // T1 = T2 * Li
    int idx = tid + j * 64, r = idx >> 5, c = idx & 31;
    float a = 0.f;
    for (int k = 0; k < RK; ++k) a += T2[r][k] * Li[k][c];
    T1[r][c] = a;
  }
  __syncthreads();
  for (int j = 0; j < 32; ++j) {                          // E = eta * Bv * T1
    int idx = tid + j * 64, p = idx >> 5, r = idx & 31;
    float a = 0.f;
    for (int k = 0; k < RK; ++k) a += Bv[p][k] * T1[k][r];
    gE[(size_t)bh * 2048 + p * RK + r] = eta * a;
  }
}

// ---------------------------------------------------------------------------
// K5: out[b,t,h,:] = (zq_norm[b,t,h,:] @ E[b,h]^T)  via f32 WMMA 16x16x4
// ---------------------------------------------------------------------------
__global__ __launch_bounds__(128)
void k5_apply(const float* __restrict__ zall, const float* __restrict__ invn,
              const float* __restrict__ gE, float* __restrict__ out)
{
  const int bh = blockIdx.x, b = bh >> 4, hh = bh & 15;
  const int lane  = threadIdx.x & 31;
  const int w     = threadIdx.x >> 5;
  const int tbase = blockIdx.y * 16;
  const int pbase = w * 16;
  const float s = invn[bh];

  const int t = tbase + (lane & 15);
  const float* zq = zall + (size_t)(b * TT + t) * NC + 512 + hh * RK;
  const float* Er = gE + (size_t)bh * 2048 + (pbase + (lane & 15)) * RK;
  const int kO = (lane >> 4) * 2;

  v8f acc = {};
#pragma unroll
  for (int kk = 0; kk < RK; kk += 4) {
    v2f a, b2;
    a.x  = zq[kk + kO] * s;   a.y  = zq[kk + kO + 1] * s;
    b2.x = Er[kk + kO];       b2.y = Er[kk + kO + 1];
    acc = __builtin_amdgcn_wmma_f32_16x16x4_f32(false, a, false, b2,
                                                (short)0, acc, false, false);
  }
  const int p  = pbase + (lane & 15);
  const int tb = tbase + (lane >> 4) * 8;
#pragma unroll
  for (int i = 0; i < 8; ++i)
    out[(((size_t)b * TT + (tb + i)) * NH + hh) * DH + p] = acc[i];
}

// ---------------------------------------------------------------------------
extern "C" void kernel_launch(void* const* d_in, const int* in_sizes, int n_in,
                              void* d_out, int out_size, void* d_ws, size_t ws_size,
                              hipStream_t stream) {
  (void)in_sizes; (void)n_in; (void)out_size; (void)ws_size;
  const float* h   = (const float*)d_in[0];
  const int*   pm  = (const int*)  d_in[1];
  const float* Wk  = (const float*)d_in[2];
  const float* Wq  = (const float*)d_in[3];
  const float* Wv  = (const float*)d_in[4];
  const float* eta = (const float*)d_in[5];
  const float* gam = (const float*)d_in[6];
  float* out = (float*)d_out;

  float* zall = (float*)d_ws;                  // 16384*2048 f32 (~128 MB)
  float* invn = zall + (size_t)MROWS * NC;     // 64
  float* gG   = invn + 64;                     // 64*1024
  float* gM   = gG + 64 * 1024;                // 64*1024
  float* gC   = gM + 64 * 1024;                // 64*2048
  float* gE   = gC + 64 * 2048;                // 64*2048
  __bf16* hb  = (__bf16*)(gE + 64 * 2048);     // 16384*1024 bf16 (32 MB)
  __bf16* wb  = hb + HTOT;                     // 2048*1024 bf16 (4 MB)

  k0_cvt    <<<(unsigned)((HTOT + WTOT) / 2048), 256, 0, stream>>>(h, Wk, Wq, Wv, hb, wb);
  k1_gemm   <<<dim3(MROWS / Bb_M, NC / Bb_N), 256, 0, stream>>>(hb, wb, zall);
  k2_maxnorm<<<64, 256, 0, stream>>>(zall, invn);
  k3_gram   <<<64, 256, 0, stream>>>(zall, pm, invn, gG, gM, gC);
  k4_solve  <<<64, 64, 0, stream>>>(gG, gM, gC, eta, gam, gE);
  k5_apply  <<<dim3(64, TT / 16), 128, 0, stream>>>(zall, invn, gE, out);
}
